// ARQS_66228395704554
// MI455X (gfx1250) — matrix operand order repaired and checked by reading, not compiled, not run on hardware
//
#include <hip/hip_runtime.h>
#include <hip/hip_bf16.h>
#include <math.h>

// ---------------------------------------------------------------------------
// Fused MLP (64->512->512->1472) + rational-quadratic-spline flow for gfx1250.
// bf16 WMMA (16x16x32) with f32 accumulation; weights pre-packed into WMMA-B
// fragment layout; all intermediates live in LDS (no 386MB param round-trip).
// ---------------------------------------------------------------------------

typedef __attribute__((ext_vector_type(16))) __bf16 v16bf;
typedef __attribute__((ext_vector_type(8)))  __bf16 v8bf;
typedef __attribute__((ext_vector_type(8)))  float  v8f;

#define TM    16      // batch rows per workgroup (one WMMA M tile)
#define HDIM  512
#define OUTD  1472    // 64 dims * 23 params
#define NB    8
#define H0S   520     // padded bf16 row stride for h0/h1 LDS tiles
#define XBS   72      // padded bf16 row stride for x tile

// ---- fragment loaders (contiguous K order, same in A and B) ----------------------
// A (16xK bf16, row-major LDS tile): lane&15 = row M; lane>>4 selects K-half.
// elements 0..7  -> K = k0 + hi*16 .. +7     (VGPR0..3)
// elements 8..15 -> K = k0 + hi*16 + 8 .. +7 (VGPR4..7)
__device__ inline v16bf load_a_frag(const __bf16* tile, int stride, int k0, int lane) {
    const int row = lane & 15;
    const int hi  = (lane >> 4) & 1;
    const __bf16* p = tile + row * stride + k0 + hi * 16;
    union { v16bf v; v8bf h[2]; } u;
    u.h[0] = *(const v8bf*)(p);        // 16B aligned -> ds_load_b128
    u.h[1] = *(const v8bf*)(p + 8);
    return u.v;
}

// B (32x16 bf16) pre-packed so each lane's 16 elements are contiguous (32B).
// lane&15 = column N; lane>=16 holds K=16..31; element e -> K = hi*16 + e.
__device__ inline v16bf load_b_frag(const __bf16* __restrict__ packed, int frag, int lane) {
    return *(const v16bf*)(packed + (size_t)frag * 512 + lane * 16);
}

__device__ inline v8f wmma_bf16(v16bf a, v16bf b, v8f c) {
    return __builtin_amdgcn_wmma_f32_16x16x32_bf16(
        /*neg_a=*/false, a, /*neg_b=*/false, b,
        /*c_mod=*/(short)0, c, /*reuse_a=*/false, /*reuse_b=*/false);
}

// C epilogue: col = lane&15 (fixed per lane!), row = r + (lane>>4)*8.
__device__ inline void store_relu_bf16(__bf16* dst, int stride, int n0, int lane,
                                       v8f acc, float bias) {
    const int lo = lane & 15;
    const int hi = lane >> 4;
    #pragma unroll
    for (int r = 0; r < 8; ++r) {
        float v = fmaxf(acc[r] + bias, 0.0f);
        dst[(r + hi * 8) * stride + n0 + lo] = (__bf16)v;
    }
}

// ---- one-time weight packer: f32 row-major (K x N) -> bf16 B-frag layout --
__global__ __launch_bounds__(256) void pack_w_kernel(const float* __restrict__ W,
                                                     __bf16* __restrict__ out,
                                                     int K, int N) {
    const int ktiles = K >> 5;
    const int total  = K * N;
    for (int idx = blockIdx.x * blockDim.x + threadIdx.x; idx < total;
         idx += gridDim.x * blockDim.x) {
        const int frag = idx >> 9;          // 512 bf16 per fragment
        const int r    = idx & 511;
        const int lane = r >> 4;
        const int e    = r & 15;
        const int kt   = frag % ktiles;
        const int nt   = frag / ktiles;
        const int k    = kt * 32 + ((lane >> 4) << 4) + e;
        const int n    = nt * 16 + (lane & 15);
        out[idx] = (__bf16)W[(size_t)k * N + n];
    }
}

// ---- fused MLP + spline ---------------------------------------------------
__global__ __launch_bounds__(256) void arqs_fused_kernel(
    const float* __restrict__ x,
    const float* __restrict__ bias0, const float* __restrict__ bias1,
    const float* __restrict__ bias2,
    const __bf16* __restrict__ W0p, const __bf16* __restrict__ W1p,
    const __bf16* __restrict__ W2p,
    float* __restrict__ z_out, float* __restrict__ ld_out) {

    extern __shared__ char smem[];
    float*  xs   = (float*)(smem);               // 16*64  f32   (4096 B)
    __bf16* xbf  = (__bf16*)(smem + 4096);       // 16*72  bf16  (2304 B)
    __bf16* h0   = (__bf16*)(smem + 6400);       // 16*520 bf16  (16640 B)
    __bf16* h1   = (__bf16*)(smem + 23040);      // 16*520 bf16  (16640 B)
    float*  par  = (float*)(smem + 39680);       // 16*1472 f32  (94208 B)
    float*  ldac = (float*)(smem + 133888);      // 16 f32

    const int tid  = threadIdx.x;
    const int lane = tid & 31;
    const int wv   = tid >> 5;
    const int rb   = blockIdx.x * TM;            // base batch row
    const int lo   = lane & 15;
    const int hi   = lane >> 4;

    // stage x tile (f32 for spline, bf16 for GEMM A)
    #pragma unroll
    for (int i = 0; i < 4; ++i) {
        const int idx = tid + i * 256;           // 0..1023
        const int r = idx >> 6, c = idx & 63;
        const float v = x[(size_t)(rb + r) * 64 + c];
        xs[r * 64 + c]  = v;
        xbf[r * XBS + c] = (__bf16)v;
    }
    if (tid < TM) ldac[tid] = 0.0f;
    __syncthreads();

    // ---------------- GEMM1: (16x64) @ (64x512) + relu -> h0 ---------------
    {
        const v16bf a0 = load_a_frag(xbf, XBS, 0, lane);
        const v16bf a1 = load_a_frag(xbf, XBS, 32, lane);
        #pragma unroll
        for (int j = 0; j < 4; ++j) {
            const int nt = wv + j * 8;           // 32 N-tiles over 8 waves
            v8f acc = {};
            acc = wmma_bf16(a0, load_b_frag(W0p, nt * 2 + 0, lane), acc);
            acc = wmma_bf16(a1, load_b_frag(W0p, nt * 2 + 1, lane), acc);
            store_relu_bf16(h0, H0S, nt * 16, lane, acc, bias0[nt * 16 + lo]);
        }
    }
    __syncthreads();

    // ---------------- GEMM2: (16x512) @ (512x512) + relu -> h1 -------------
    {
        v8f acc[4] = {{}, {}, {}, {}};
        for (int k = 0; k < 16; ++k) {
            const v16bf a = load_a_frag(h0, H0S, k * 32, lane);
            #pragma unroll
            for (int j = 0; j < 4; ++j) {
                const int nt = wv + j * 8;
                acc[j] = wmma_bf16(a, load_b_frag(W1p, nt * 16 + k, lane), acc[j]);
            }
        }
        #pragma unroll
        for (int j = 0; j < 4; ++j) {
            const int nt = wv + j * 8;
            store_relu_bf16(h1, H0S, nt * 16, lane, acc[j], bias1[nt * 16 + lo]);
        }
    }
    __syncthreads();

    // ---------------- GEMM3: (16x512) @ (512x1472) + bias -> par (f32) -----
    for (int nt = wv; nt < 92; nt += 8) {
        v8f acc = {};
        for (int k = 0; k < 16; ++k) {
            const v16bf a = load_a_frag(h1, H0S, k * 32, lane);
            acc = wmma_bf16(a, load_b_frag(W2p, nt * 16 + k, lane), acc);
        }
        const int n0 = nt * 16;
        const float bias = bias2[n0 + lo];
        #pragma unroll
        for (int r = 0; r < 8; ++r)
            par[(r + hi * 8) * OUTD + n0 + lo] = acc[r] + bias;
    }
    __syncthreads();

    // ---------------- rational-quadratic spline per (row, dim) -------------
    #pragma unroll
    for (int i = 0; i < 4; ++i) {
        const int item = tid + i * 256;          // 0..1023
        const int row = item >> 6;
        const int d   = item & 63;
        const float* p = par + row * OUTD + d * 23;
        const float xin = fminf(fmaxf(xs[row * 64 + d], 0.0f), 1.0f);

        // softmax widths / heights (+min-bin affine)
        float wb[NB], hb[NB];
        float mw = -3.0e38f, mh = -3.0e38f;
        #pragma unroll
        for (int b = 0; b < NB; ++b) { mw = fmaxf(mw, p[b]); mh = fmaxf(mh, p[NB + b]); }
        float sw = 0.0f, sh = 0.0f;
        #pragma unroll
        for (int b = 0; b < NB; ++b) {
            wb[b] = __expf(p[b] - mw);      sw += wb[b];
            hb[b] = __expf(p[NB + b] - mh); sh += hb[b];
        }
        const float iw = (1.0f - 0.001f * NB) / sw;
        const float ih = (1.0f - 0.001f * NB) / sh;
        #pragma unroll
        for (int b = 0; b < NB; ++b) {
            wb[b] = 0.001f + wb[b] * iw;
            hb[b] = 0.001f + hb[b] * ih;
        }
        // derivatives: softplus + pad with 1.0 at both ends
        float dv[NB + 1];
        dv[0] = 1.0f; dv[NB] = 1.0f;
        #pragma unroll
        for (int j = 0; j < NB - 1; ++j) {
            const float t = p[2 * NB + j];
            dv[j + 1] = ((t > 20.0f) ? t : log1pf(__expf(t))) + 0.001f;
        }
        // bin search: count knots (0, cumsum(w)) strictly below xin
        float kx = 0.0f; int cnt = 0;
        #pragma unroll
        for (int b = 0; b < NB; ++b) { if (kx < xin) cnt++; kx += wb[b]; }
        if (kx < xin) cnt++;
        const int kbin = min(max(cnt, 1), NB) - 1;   // 0..7
        // prefix sums up to kbin + gather (predicated selects, no scratch)
        float x_k = 0.0f, y_k = 0.0f, w_k = 0.0f, h_k = 0.0f;
        #pragma unroll
        for (int b = 0; b < NB; ++b) {
            if (b < kbin)  { x_k += wb[b]; y_k += hb[b]; }
            if (b == kbin) { w_k = wb[b];  h_k = hb[b]; }
        }
        float d_k = 1.0f, d_k1 = 1.0f;
        #pragma unroll
        for (int j = 0; j <= NB; ++j) {
            if (j == kbin)     d_k  = dv[j];
            if (j == kbin + 1) d_k1 = dv[j];
        }
        const float s_k = h_k / w_k;
        const float th  = (xin - x_k) / w_k;
        const float omt = 1.0f - th;
        const float t1t = th * omt;
        const float den = s_k + (d_k + d_k1 - 2.0f * s_k) * t1t;
        const float zv  = y_k + h_k * (s_k * th * th + d_k * t1t) / (den + 1e-6f);
        const float nom = s_k * s_k * (d_k1 * th * th + 2.0f * s_k * t1t + d_k * omt * omt);
        const float ld  = __logf(nom / (den * den + 1e-6f));

        z_out[(size_t)(rb + row) * 64 + d] = zv;
        atomicAdd(&ldac[row], ld);               // ds_add_f32
    }
    __syncthreads();
    if (tid < TM) ld_out[rb + tid] = ldac[tid];
}

// ---------------------------------------------------------------------------
extern "C" void kernel_launch(void* const* d_in, const int* in_sizes, int n_in,
                              void* d_out, int out_size, void* d_ws, size_t ws_size,
                              hipStream_t stream) {
    const float* x  = (const float*)d_in[0];
    const float* W0 = (const float*)d_in[1];
    const float* b0 = (const float*)d_in[2];
    const float* W1 = (const float*)d_in[3];
    const float* b1 = (const float*)d_in[4];
    const float* W2 = (const float*)d_in[5];
    const float* b2 = (const float*)d_in[6];
    // num_bins (d_in[7]) fixed at 8 per reference setup

    const int B = in_sizes[0] / 64;   // 65536

    // workspace: packed bf16 weights (2 MB total)
    __bf16* W0p = (__bf16*)d_ws;                                   //  64x512
    __bf16* W1p = (__bf16*)((char*)d_ws + 64 * 512 * 2);           // 512x512
    __bf16* W2p = (__bf16*)((char*)d_ws + 64 * 512 * 2 + 512 * 512 * 2); // 512x1472

    pack_w_kernel<<<(64 * 512 + 255) / 256, 256, 0, stream>>>(W0, W0p, 64, 512);
    pack_w_kernel<<<(512 * 512 + 255) / 256, 256, 0, stream>>>(W1, W1p, 512, 512);
    pack_w_kernel<<<(512 * 1472 + 255) / 256, 256, 0, stream>>>(W2, W2p, 512, 1472);

    float* z  = (float*)d_out;
    float* ld = z + (size_t)B * 64;

    const size_t shmem = 133952;   // x(f32+bf16) + h0 + h1 + params + logdet
    arqs_fused_kernel<<<B / TM, 256, shmem, stream>>>(x, b0, b1, b2,
                                                      W0p, W1p, W2p, z, ld);
}
